// SocialLSTM_76742475644942
// MI455X (gfx1250) — matrix operand order repaired and checked by reading, not compiled, not run on hardware
//
#include <hip/hip_runtime.h>
#include <hip/hip_bf16.h>
#include <stdint.h>

// ---------------------------------------------------------------------------
// Social-LSTM forward for gfx1250 (MI455X). Mixed precision:
//   - LSTM state / nonlinearities in f32
//   - GEMM operands in bf16, f32 accumulation via v_wmma_f32_16x16x32_bf16
//   - A-tiles staged to LDS with global_load_async_to_lds_b128 (ASYNCcnt)
// ---------------------------------------------------------------------------

typedef __attribute__((ext_vector_type(16))) __bf16 v16bf;
typedef __attribute__((ext_vector_type(8)))  float  v8f;

#define N_AG   1024
#define HID    64
#define KSOC   4096          // 8*8 cells * 64 hidden
#define NSOC   256
#define KZ     384           // 64 (r) + 256 (e) + 64 (h)
#define NGATE  256

// float -> bf16 round-to-nearest-even (bit-exact, no ISA surprises)
__device__ __forceinline__ unsigned short f2bf(float f) {
    union { float f; uint32_t u; } c; c.f = f;
    uint32_t u = c.u;
    uint32_t r = u + 0x7fffu + ((u >> 16) & 1u);
    if ((u & 0x7f800000u) == 0x7f800000u) r = u;   // inf/nan passthrough
    return (unsigned short)(r >> 16);
}

// generic (shared) pointer -> 32-bit LDS byte offset for async-LDS VDST
__device__ __forceinline__ unsigned lds_offset(const void* p) {
    return (unsigned)(uintptr_t)(__attribute__((address_space(3))) const void*)p;
}

// ---------------------------------------------------------------------------
// One-time (per launch) weight conversion + state init.
// ---------------------------------------------------------------------------
__global__ void sl_init_kernel(const float* __restrict__ Wsoc,
                               const float* __restrict__ Wih,
                               const float* __restrict__ Whh,
                               const float* __restrict__ bih,
                               const float* __restrict__ bhh,
                               const float* __restrict__ h0,
                               const float* __restrict__ c0,
                               unsigned short* __restrict__ WsocB,
                               unsigned short* __restrict__ WzB,
                               float* __restrict__ bz,
                               float* __restrict__ h,
                               float* __restrict__ c) {
    int idx = blockIdx.x * blockDim.x + threadIdx.x;
    int stride = gridDim.x * blockDim.x;
    for (int i = idx; i < KSOC * NSOC; i += stride) WsocB[i] = f2bf(Wsoc[i]);
    for (int i = idx; i < 320 * NGATE; i += stride) WzB[i] = f2bf(Wih[i]);
    for (int i = idx; i < HID * NGATE; i += stride) WzB[320 * NGATE + i] = f2bf(Whh[i]);
    for (int i = idx; i < NGATE; i += stride) bz[i] = bih[i] + bhh[i];
    for (int i = idx; i < N_AG * HID; i += stride) { h[i] = h0[i]; c[i] = c0[i]; }
}

// ---------------------------------------------------------------------------
// Social pooling: one workgroup per agent i. Accumulate neighbor h into a
// 64-cell x 64-hidden LDS grid with ds_add_f32, emit bf16 row of H, and fill
// z[i] = [ relu(coords@W_emb+b) | (e filled later) | bf16(h[i]) ].
// ---------------------------------------------------------------------------
__global__ __launch_bounds__(256)
void sl_pool_kernel(const float* __restrict__ X,
                    const int* __restrict__ masks,
                    const float* __restrict__ h,
                    const float* __restrict__ Wemb,
                    const float* __restrict__ bemb,
                    unsigned short* __restrict__ Hb,
                    unsigned short* __restrict__ z,
                    int t) {
    __shared__ float Hacc[KSOC];
    const int i = blockIdx.x;
    const int tid = threadIdx.x;
    for (int k = tid; k < KSOC; k += 256) Hacc[k] = 0.f;

    const float* xt = X + (size_t)t * N_AG * 4;
    const float cix = xt[i * 4 + 2];
    const float ciy = xt[i * 4 + 3];
    const int mi = masks[t * N_AG + i];
    __syncthreads();

    if (mi != 0) {
        for (int j = tid; j < N_AG; j += 256) {
            if (j == i) continue;
            if (masks[t * N_AG + j] == 0) continue;
            // rel = coords[j] - coords[i]; CELL == 1.0
            float rx = xt[j * 4 + 2] - cix;
            float ry = xt[j * 4 + 3] - ciy;
            int gx = (int)truncf(rx);
            int gy = (int)truncf(ry);
            if (gx < -3 || gx > 3 || gy < -3 || gy > 3) continue;  // |g| <= half-1
            int cell = (gx + 4) * 8 + (gy + 4);
            const float* hj = h + (size_t)j * HID;
            float* dst = &Hacc[cell * HID];
            #pragma unroll 8
            for (int k = 0; k < HID; ++k) atomicAdd(&dst[k], hj[k]);  // ds_add_f32
        }
    }
    __syncthreads();

    unsigned short* Hrow = Hb + (size_t)i * KSOC;
    for (int k = tid; k < KSOC; k += 256) Hrow[k] = f2bf(Hacc[k]);

    if (tid < HID) {
        float r = cix * Wemb[tid] + ciy * Wemb[HID + tid] + bemb[tid];
        r = r > 0.f ? r : 0.f;
        z[(size_t)i * KZ + tid] = f2bf(r);
        z[(size_t)i * KZ + 320 + tid] = f2bf(h[(size_t)i * HID + tid]);
    }
}

// ---------------------------------------------------------------------------
// bf16 GEMM: C[M x Ncols] = act(A[M x K] @ B[K x Ncols] + bias)
// Block tile 64x128, BK=32, 256 threads = 8 waves.
// Wave layout: 4 row-tiles x 2 col-groups; each wave = 1 A-frag x 4 B-frags
// -> 4 x v_wmma_f32_16x16x32_bf16 per K-chunk (operand reuse on A).
// A-tile staged via global_load_async_to_lds_b128; B-tile transposed in-flight.
// Output: f32 (Cf) or relu->bf16 (Cb) at column offset col0, row stride ldc.
// ---------------------------------------------------------------------------
__global__ __launch_bounds__(256)
void sl_gemm_bf16_kernel(const unsigned short* __restrict__ A, int lda,
                         const unsigned short* __restrict__ B, int ldb,
                         const float* __restrict__ bias, int K,
                         float* __restrict__ Cf,
                         unsigned short* __restrict__ Cb,
                         int ldc, int col0, int relu) {
    __shared__ __align__(16) unsigned short As[64 * 32];    // [m][k]      4 KB
    __shared__ __align__(16) unsigned short Bs[128 * 32];   // [n][k] (T)  8 KB

    const int tid  = threadIdx.x;
    const int lane = tid & 31;
    const int wave = tid >> 5;
    const int m0   = blockIdx.y * 64;
    const int n0   = blockIdx.x * 128;
    const int wm   = (wave & 3) * 16;      // wave row-tile offset
    const int wn   = (wave >> 2) * 64;     // wave col group (4 tiles of 16)
    const int half = lane >> 4;            // 0: lanes 0-15, 1: lanes 16-31
    const int lr   = lane & 15;

    v8f acc[4] = {v8f{}, v8f{}, v8f{}, v8f{}};

    const int arow = tid >> 2,  acol = (tid & 3) * 8;    // A tile 64x32
    const int brow = tid >> 4,  bcol = (tid & 15) * 8;   // B tile 32x128, 2 rows/thr

    const unsigned aoff = lds_offset(&As[arow * 32 + acol]);

    for (int k0 = 0; k0 < K; k0 += 32) {
        // --- A tile: async copy global -> LDS (no VGPR round-trip) ---
        unsigned long long ga =
            (unsigned long long)(uintptr_t)(A + (size_t)(m0 + arow) * lda + k0 + acol);
        asm volatile("global_load_async_to_lds_b128 %0, %1, off"
                     :: "v"(aoff), "v"(ga) : "memory");

        // --- B tile: load + transpose into [n][k] layout ---
        #pragma unroll
        for (int rr = 0; rr < 2; ++rr) {
            int row = brow + rr * 16;
            union { uint4 q; unsigned short u[8]; } bv;
            bv.q = *(const uint4*)(B + (size_t)(k0 + row) * ldb + n0 + bcol);
            #pragma unroll
            for (int jj = 0; jj < 8; ++jj) Bs[(bcol + jj) * 32 + row] = bv.u[jj];
        }

        if (k0 + 32 < K) {  // hint next K-tile toward the caches
            __builtin_prefetch(B + (size_t)(k0 + 32 + brow) * ldb + n0 + bcol, 0, 1);
        }

        asm volatile("s_wait_asynccnt 0x0" ::: "memory");
        __syncthreads();

        // --- fragments per documented gfx1250 VGPR layouts ---
        union { v16bf v; uint4 q[2]; } af, bf[4];
        // A 16x32: lane half 0 -> K 0-7 & 16-23 ; half 1 -> K 8-15 & 24-31
        af.q[0] = *(const uint4*)&As[(wm + lr) * 32 + half * 8];
        af.q[1] = *(const uint4*)&As[(wm + lr) * 32 + 16 + half * 8];
        // B 32x16: lane = column, half 0 -> K 0-7 & 16-23 ; half 1 -> K 8-15 & 24-31
        #pragma unroll
        for (int tt = 0; tt < 4; ++tt) {
            const unsigned short* bp = &Bs[(wn + tt * 16 + lr) * 32 + half * 8];
            bf[tt].q[0] = *(const uint4*)bp;
            bf[tt].q[1] = *(const uint4*)(bp + 16);
        }
        #pragma unroll
        for (int tt = 0; tt < 4; ++tt) {
            acc[tt] = __builtin_amdgcn_wmma_f32_16x16x32_bf16(
                false, af.v, false, bf[tt].v, (short)0, acc[tt], false, false);
        }
        __syncthreads();
    }

    // --- epilogue: C/D layout row = v + 8*half, col = lane%16 ---
    #pragma unroll
    for (int tt = 0; tt < 4; ++tt) {
        int ncol = n0 + wn + tt * 16 + lr;
        float bb = bias[ncol];
        #pragma unroll
        for (int v = 0; v < 8; ++v) {
            int mrow = m0 + wm + v + half * 8;
            float val = acc[tt][v] + bb;
            if (relu) val = val > 0.f ? val : 0.f;
            if (Cf) Cf[(size_t)mrow * ldc + col0 + ncol] = val;
            else    Cb[(size_t)mrow * ldc + col0 + ncol] = f2bf(val);
        }
    }
}

// ---------------------------------------------------------------------------
// LSTM cell + output head + masking. 256 threads = 4 agents x 64 hidden.
// ---------------------------------------------------------------------------
__global__ __launch_bounds__(256)
void sl_cell_kernel(const float* __restrict__ gates,
                    const int* __restrict__ masks,
                    const float* __restrict__ Y,
                    const float* __restrict__ Wout,
                    const float* __restrict__ bout,
                    const int* __restrict__ Tpred,
                    float* __restrict__ h,
                    float* __restrict__ c,
                    float* __restrict__ out,
                    int t) {
    __shared__ float hn[4][HID];
    const int sub = threadIdx.x >> 6;
    const int k   = threadIdx.x & 63;
    const int a   = blockIdx.x * 4 + sub;
    const int active = (t <= Tpred[0]);

    const float* g = gates + (size_t)a * NGATE;
    float ig = g[k], fg = g[64 + k], gg = g[128 + k], og = g[192 + k];
    float si = 1.f / (1.f + __expf(-ig));
    float sf = 1.f / (1.f + __expf(-fg));
    float so = 1.f / (1.f + __expf(-og));
    float cn = sf * c[(size_t)a * HID + k] + si * tanhf(gg);
    float hv = so * tanhf(cn);
    hn[sub][k] = hv;
    if (active) {
        c[(size_t)a * HID + k] = cn;
        h[(size_t)a * HID + k] = hv;
    }
    __syncthreads();

    if (k < 2) {
        float s = bout[k];
        #pragma unroll 8
        for (int j = 0; j < HID; ++j) s += hn[sub][j] * Wout[j * 2 + k];
        int mi = masks[t * N_AG + a];
        s *= (mi != 0) ? 1.f : 0.f;
        if (t > 3) {
            int m3 = masks[(t - 3) * N_AG + a];
            if (mi != 0 && m3 == 0) s = Y[((size_t)t * N_AG + a) * 2 + k];
        }
        if (!active) s = 0.f;
        out[((size_t)t * N_AG + a) * 2 + k] = s;
    }
}

// ---------------------------------------------------------------------------
// Host-side orchestration (graph-capture safe: only kernel launches on stream)
// ---------------------------------------------------------------------------
extern "C" void kernel_launch(void* const* d_in, const int* in_sizes, int n_in,
                              void* d_out, int out_size, void* d_ws, size_t ws_size,
                              hipStream_t stream) {
    const float* X      = (const float*)d_in[0];
    const int*   pmask  = (const int*)  d_in[1];
    const float* h0     = (const float*)d_in[2];
    const float* c0     = (const float*)d_in[3];
    const float* Y      = (const float*)d_in[4];
    const int*   Tpred  = (const int*)  d_in[6];   // d_in[5] = T_obs (unused)
    const float* Wemb   = (const float*)d_in[7];
    const float* bemb   = (const float*)d_in[8];
    const float* Wsoc   = (const float*)d_in[9];
    const float* bsoc   = (const float*)d_in[10];
    const float* Wih    = (const float*)d_in[11];
    const float* bih    = (const float*)d_in[12];
    const float* Whh    = (const float*)d_in[13];
    const float* bhh    = (const float*)d_in[14];
    const float* Wout   = (const float*)d_in[15];
    const float* bout   = (const float*)d_in[16];

    const int T = in_sizes[0] / (N_AG * 4);

    // workspace layout (256B-aligned offsets), ~12.5 MB total
    char* ws = (char*)d_ws;
    unsigned short* WsocB = (unsigned short*)(ws);                       // 2 MB
    unsigned short* WzB   = (unsigned short*)(ws + 2097152);             // 192 KB
    float*          bz    = (float*)         (ws + 2293760);             // 1 KB
    float*          hbuf  = (float*)         (ws + 2294784);             // 256 KB
    float*          cbuf  = (float*)         (ws + 2556928);             // 256 KB
    unsigned short* Hb    = (unsigned short*)(ws + 2819072);             // 8 MB
    unsigned short* zb    = (unsigned short*)(ws + 11207680);            // 768 KB
    float*          gbuf  = (float*)         (ws + 11994112);            // 1 MB

    sl_init_kernel<<<1024, 256, 0, stream>>>(Wsoc, Wih, Whh, bih, bhh, h0, c0,
                                             WsocB, WzB, bz, hbuf, cbuf);

    for (int t = 0; t < T; ++t) {
        // 1) social pooling + r + h staging (LDS f32 atomics, bf16 out)
        sl_pool_kernel<<<N_AG, 256, 0, stream>>>(X, pmask, hbuf, Wemb, bemb,
                                                 Hb, zb, t);
        // 2) e = relu(H @ W_soc + b_soc)  -> z[:, 64:320] (bf16)
        sl_gemm_bf16_kernel<<<dim3(NSOC / 128, N_AG / 64), 256, 0, stream>>>(
            Hb, KSOC, WsocB, NSOC, bsoc, KSOC,
            nullptr, zb, KZ, /*col0=*/64, /*relu=*/1);
        // 3) gates = [r|e|h] @ [W_ih;W_hh] + (b_ih+b_hh)  -> f32
        sl_gemm_bf16_kernel<<<dim3(NGATE / 128, N_AG / 64), 256, 0, stream>>>(
            zb, KZ, WzB, NGATE, bz, KZ,
            gbuf, nullptr, NGATE, /*col0=*/0, /*relu=*/0);
        // 4) LSTM cell + output head + masking
        sl_cell_kernel<<<N_AG / 4, 256, 0, stream>>>(gbuf, pmask, Y, Wout, bout,
                                                     Tpred, hbuf, cbuf,
                                                     (float*)d_out, t);
    }
}
